// Attention_50646254355243
// MI455X (gfx1250) — matrix operand order repaired and checked by reading, not compiled, not run on hardware
//
#include <hip/hip_runtime.h>

// ---------------------------------------------------------------------------
// MI455X (gfx1250) attention block: bf16 pre-pass -> QKV proj + RMSNorm + RoPE
// -> flash attention -> output projection.  All matmuls via
// v_wmma_f32_16x16x32_bf16 (wave32).  Double-buffered LDS tiles: TDM
// (tensor_load_to_lds) streams the A-side tiles while waves run WMMAs;
// cooperative tiles are staged in VGPRs during compute and stored after.
// Shapes: B=2, T=2048, HIDDEN=2048, NH=32, NKV=8, D=64, group=4.
// ---------------------------------------------------------------------------

typedef __attribute__((ext_vector_type(16))) __bf16        v16bf;
typedef __attribute__((ext_vector_type(8)))  float         v8f;
typedef __attribute__((ext_vector_type(4)))  unsigned int  u32x4;
typedef __attribute__((ext_vector_type(8)))  int           i32x8;
typedef __attribute__((ext_vector_type(4)))  int           i32x4;

#if defined(__has_builtin)
#  if __has_builtin(__builtin_amdgcn_tensor_load_to_lds) && \
      __has_builtin(__builtin_amdgcn_s_wait_tensorcnt)
#    define USE_TDM 1
#  endif
#endif
#ifndef USE_TDM
#  define USE_TDM 0
#endif

__device__ __forceinline__ unsigned short f2bf(float f) {
    unsigned u = __builtin_bit_cast(unsigned, f);
    u += 0x7FFFu + ((u >> 16) & 1u);          // round-to-nearest-even
    return (unsigned short)(u >> 16);
}

// Two contiguous 16B LDS reads -> one 16-element bf16 fragment.
__device__ __forceinline__ v16bf ld_frag2(const unsigned short* p0,
                                          const unsigned short* p1) {
    union { v16bf v; uint4 q[2]; } u;
    u.q[0] = *(const uint4*)p0;
    u.q[1] = *(const uint4*)p1;
    return u.v;
}
// A-fragment 16x32: lane half=0 -> K{0..7,16..23}, half=1 -> K{8..15,24..31}
__device__ __forceinline__ v16bf a_frag(const unsigned short* row, int kc, int half) {
    return ld_frag2(row + kc + half * 8, row + kc + 16 + half * 8);
}
// B-fragment 32x16 from K-contiguous row (B^T storage): half=0 -> K{0..7,16..23}, half=1 -> K{8..15,24..31}
__device__ __forceinline__ v16bf b_frag(const unsigned short* row, int kc, int half) {
    return ld_frag2(row + kc + half * 8, row + kc + 16 + half * 8);
}

#define WMMA_BF16(ACC, A, B)                                                     \
    (ACC) = __builtin_amdgcn_wmma_f32_16x16x32_bf16(false, (A), false, (B),      \
                                                    (short)0, (ACC), false, false)

// ---------------------------------------------------------------------------
// TDM: 2D tile load (tile width fixed: 64 bf16 = 128B rows) with LDS padding
// of 16B per 128B row => padded row stride 144B (stride-72 ushort).
// ---------------------------------------------------------------------------
#if USE_TDM
__device__ __forceinline__ i32x8 tdm_desc2d(unsigned tile_h, unsigned stride_elems) {
    i32x8 g1;
    g1[0] = (int)(0x00010000u        // data_size = 2B
                  | (1u << 20)       // pad_enable
                  | (4u << 22)       // pad_interval: every 32 DWORDs (128B)
                  | (3u << 25));     // pad_amount: 4 DWORDs (16B)
    g1[1] = (int)((stride_elems & 0xFFFFu) << 16);   // tensor_dim0 lo
    g1[2] = (int)(stride_elems >> 16);               // tensor_dim0 hi, dim1 lo=0
    g1[3] = (int)(16u | (64u << 16));                // dim1 hi (1<<20), tile_dim0=64
    g1[4] = (int)tile_h;                             // tile_dim1
    g1[5] = (int)stride_elems;                       // tensor_dim0_stride lo
    g1[6] = 0; g1[7] = 0;
    return g1;
}
__device__ __forceinline__ void tdm_issue(unsigned lds_off, const void* g, i32x8 g1) {
    unsigned long long ga = (unsigned long long)(size_t)g;
    u32x4 g0;
    g0.x = 1u;                                        // count=1
    g0.y = lds_off;                                   // lds_addr (bytes)
    g0.z = (unsigned)ga;                              // global_addr lo
    g0.w = (unsigned)((ga >> 32) & 0x01FFFFFFu) | 0x80000000u;  // addr hi | type=2
    i32x4 z4 = {};
#if __clang_major__ >= 23
    i32x8 z8 = {};
    __builtin_amdgcn_tensor_load_to_lds(g0, g1, z4, z4, z8, 0);
#else
    __builtin_amdgcn_tensor_load_to_lds(g0, g1, z4, z4, 0);
#endif
}
__device__ __forceinline__ unsigned lds_off_of(const void* p) {
    return (unsigned)(size_t)p;                       // low 32 bits = LDS offset
}
#endif

// ---------------------------------------------------------------------------
// Pre-pass: fp32 -> bf16 elementwise (x), and convert+transpose (weights).
// ---------------------------------------------------------------------------
__global__ __launch_bounds__(256) void cvt_bf16_kernel(
    const float* __restrict__ in, unsigned short* __restrict__ out, int n4)
{
    const int i = (blockIdx.x * 256 + threadIdx.x) * 4;
    if (i < n4 * 4) {
        float4 v = *(const float4*)(in + i);
        ushort4 o;
        o.x = f2bf(v.x); o.y = f2bf(v.y); o.z = f2bf(v.z); o.w = f2bf(v.w);
        *(ushort4*)(out + i) = o;
    }
}

__global__ __launch_bounds__(256) void transpose_bf16_kernel(
    const float* __restrict__ in, unsigned short* __restrict__ outT, int K, int N)
{   // in: K x N (f32) -> outT: N x K (bf16)
    __shared__ unsigned short tile[64][65];
    const int k0 = blockIdx.y * 64, n0 = blockIdx.x * 64;
    const int tid = threadIdx.x;
#pragma unroll
    for (int i = 0; i < 16; ++i) {
        int e = tid * 16 + i, r = e >> 6, c = e & 63;
        tile[r][c] = f2bf(in[(size_t)(k0 + r) * N + n0 + c]);
    }
    __syncthreads();
#pragma unroll
    for (int i = 0; i < 16; ++i) {
        int e = tid * 16 + i, r = e >> 6, c = e & 63;     // r=n-local, c=k-local
        outT[(size_t)(n0 + r) * K + k0 + c] = tile[c][r];
    }
}

// ---------------------------------------------------------------------------
// Kernel A: QKV projection (bf16 WMMA) + RMSNorm + RoPE, bf16 outputs.
// Grid: (48 col-tiles of 64, BT/128).  Block 256 (8 waves).  Double-buffered.
// ---------------------------------------------------------------------------
__global__ __launch_bounds__(256) void qkv_rope_kernel(
    const unsigned short* __restrict__ xb,
    const unsigned short* __restrict__ WqT, const unsigned short* __restrict__ WkT,
    const unsigned short* __restrict__ WvT,
    const float* __restrict__ qlnw, const float* __restrict__ klnw,
    unsigned short* __restrict__ Qb, unsigned short* __restrict__ Kb,
    unsigned short* __restrict__ Vb)
{
    __shared__ __align__(16) unsigned short lx[2][128][72]; // x tiles 128x64
    __shared__ __align__(16) unsigned short lw[2][64][72];  // W^T tiles 64x64
    __shared__ float lt[128][65];

    const int tid  = threadIdx.x;
    const int lane = tid & 31, wave = tid >> 5;
    const int half = lane >> 4, ln = lane & 15;
    const int r0   = blockIdx.y * 128;
    const int ct   = blockIdx.x;

    const unsigned short* WT; int n0, kind;      // kind: 0=q 1=k 2=v
    if (ct < 32)      { WT = WqT; n0 = ct * 64;        kind = 0; }
    else if (ct < 40) { WT = WkT; n0 = (ct - 32) * 64; kind = 1; }
    else              { WT = WvT; n0 = (ct - 40) * 64; kind = 2; }

    const int rowW = tid >> 2, c0W = (tid & 3) * 16;      // W stage coords
    v8f acc[4] = {};
    uint4 w0, w1;
#if USE_TDM
    const i32x8 dx = tdm_desc2d(128u, 2048u);
#else
    const int rowX = tid >> 1, c0X = (tid & 1) * 32;
    uint4 xr[4];
#endif

    // ---- prologue: tile 0 ----
#if USE_TDM
    if (wave == 0) tdm_issue(lds_off_of(&lx[0][0][0]), xb + (size_t)r0 * 2048, dx);
#else
    {
        const unsigned short* s = xb + (size_t)(r0 + rowX) * 2048 + c0X;
#pragma unroll
        for (int q = 0; q < 4; ++q) *(uint4*)&lx[0][rowX][c0X + q * 8] = *(const uint4*)(s + q * 8);
    }
#endif
    {
        const unsigned short* s = WT + (size_t)(n0 + rowW) * 2048 + c0W;
        *(uint4*)&lw[0][rowW][c0W]     = *(const uint4*)(s);
        *(uint4*)&lw[0][rowW][c0W + 8] = *(const uint4*)(s + 8);
    }

    for (int i = 0; i < 32; ++i) {
        const int cur = i & 1, nxt = cur ^ 1;
#if USE_TDM
        if (wave == 0) __builtin_amdgcn_s_wait_tensorcnt(0);
#endif
        __syncthreads();

        const bool more = (i + 1) < 32;
        const int  kcn  = (i + 1) * 64;
        if (more) {                           // start next tile (overlaps compute)
#if USE_TDM
            if (wave == 0)
                tdm_issue(lds_off_of(&lx[nxt][0][0]), xb + (size_t)r0 * 2048 + kcn, dx);
#else
            const unsigned short* s = xb + (size_t)(r0 + rowX) * 2048 + kcn + c0X;
#pragma unroll
            for (int q = 0; q < 4; ++q) xr[q] = *(const uint4*)(s + q * 8);
#endif
            const unsigned short* sw = WT + (size_t)(n0 + rowW) * 2048 + kcn + c0W;
            w0 = *(const uint4*)(sw);
            w1 = *(const uint4*)(sw + 8);
        }

        // ---- compute on cur ----
        const unsigned short* arow = &lx[cur][wave * 16 + ln][0];
#pragma unroll
        for (int kk = 0; kk < 64; kk += 32) {
            v16bf a  = a_frag(arow, kk, half);
            v16bf b0 = b_frag(&lw[cur][ln][0],      kk, half);
            v16bf b1 = b_frag(&lw[cur][16 + ln][0], kk, half);
            v16bf b2 = b_frag(&lw[cur][32 + ln][0], kk, half);
            v16bf b3 = b_frag(&lw[cur][48 + ln][0], kk, half);
            WMMA_BF16(acc[0], a, b0); WMMA_BF16(acc[1], a, b1);
            WMMA_BF16(acc[2], a, b2); WMMA_BF16(acc[3], a, b3);
        }

        if (more) {                           // commit staged regs to next buffer
#if !USE_TDM
#pragma unroll
            for (int q = 0; q < 4; ++q) *(uint4*)&lx[nxt][rowX][c0X + q * 8] = xr[q];
#endif
            *(uint4*)&lw[nxt][rowW][c0W]     = w0;
            *(uint4*)&lw[nxt][rowW][c0W + 8] = w1;
        }
    }

#pragma unroll
    for (int t = 0; t < 4; ++t)
#pragma unroll
        for (int r = 0; r < 8; ++r)
            lt[wave * 16 + r + 8 * half][t * 16 + ln] = acc[t][r];
    __syncthreads();

    if (tid < 128) {                          // row-wise RMSNorm + RoPE epilogue
        const int row = tid, grow = r0 + row;
        const int b = grow >> 11, tpos = grow & 2047;
        float v[64];
#pragma unroll
        for (int i = 0; i < 64; ++i) v[i] = lt[row][i];

        if (kind < 2) {
            float ss = 0.f;
#pragma unroll
            for (int i = 0; i < 64; ++i) ss += v[i] * v[i];
            const float inv = rsqrtf(ss * (1.f / 64.f) + 1e-5f);
            const float* lnw = (kind == 0) ? qlnw : klnw;
#pragma unroll
            for (int i = 0; i < 64; ++i) v[i] *= inv * lnw[i];
#pragma unroll
            for (int i = 0; i < 32; ++i) {    // RoPE, theta=1e6, d=64
                const float fr = __expf(-13.815510557964274f *
                                        ((float)(2 * i) * (1.f / 64.f)));
                float s, c; __sincosf((float)tpos * fr, &s, &c);
                const float x1 = v[i], x2 = v[i + 32];
                v[i]      = x1 * c - x2 * s;
                v[i + 32] = x2 * c + x1 * s;
            }
        }
        unsigned short* dst;
        if (kind == 0)      dst = Qb + ((size_t)(b * 32 + ct)       * 2048 + tpos) * 64;
        else if (kind == 1) dst = Kb + ((size_t)(b * 8 + (ct - 32)) * 2048 + tpos) * 64;
        else                dst = Vb + ((size_t)(b * 8 + (ct - 40)) * 2048 + tpos) * 64;
#pragma unroll
        for (int i = 0; i < 64; ++i) dst[i] = f2bf(v[i]);
    }
}

// ---------------------------------------------------------------------------
// Kernel B: flash attention (online softmax), causal, GQA group=4.
// Grid: (T/128, NH, B). Block 256: 8 waves x (16 q-rows, 64 cols).
// K tiles via TDM, V tiles staged+transposed; both double-buffered.
// ---------------------------------------------------------------------------
__global__ __launch_bounds__(256) void flash_attn_kernel(
    const unsigned short* __restrict__ Qb, const unsigned short* __restrict__ Kb,
    const unsigned short* __restrict__ Vb, unsigned short* __restrict__ AO)
{
    __shared__ __align__(16) unsigned short lq[128][72];
    __shared__ __align__(16) unsigned short lk[2][64][72];
    __shared__ __align__(16) unsigned short lvT[2][64][72]; // V^T [d][t]
    __shared__ __align__(16) unsigned short lp[128][72];    // P stripes

    const int tid  = threadIdx.x;
    const int lane = tid & 31, wave = tid >> 5;
    const int half = lane >> 4, ln = lane & 15;
    const int q0 = blockIdx.x * 128;
    const int h  = blockIdx.y, b = blockIdx.z;
    const int kv = h >> 2;

    const unsigned short* Qh = Qb + (size_t)(b * 32 + h) * 2048 * 64;
    const unsigned short* Kh = Kb + (size_t)(b * 8 + kv) * 2048 * 64;
    const unsigned short* Vh = Vb + (size_t)(b * 8 + kv) * 2048 * 64;

    const int rowS = tid >> 2, c0S = (tid & 3) * 16;  // K/V stage coords
    uint4 vr0, vr1;
#if !USE_TDM
    uint4 kr0, kr1;
#endif

    // ---- prologue: Q tile + K/V tile 0 ----
#if USE_TDM
    if (wave == 0) {
        tdm_issue(lds_off_of(&lq[0][0]),       Qh + (size_t)q0 * 64, tdm_desc2d(128u, 64u));
        tdm_issue(lds_off_of(&lk[0][0][0]),    Kh,                   tdm_desc2d(64u, 64u));
    }
#else
    {   // cooperative Q tile 128x64
        const int row = tid >> 1, c0 = (tid & 1) * 32;
        const unsigned short* s = Qh + (size_t)(q0 + row) * 64 + c0;
#pragma unroll
        for (int q = 0; q < 4; ++q) *(uint4*)&lq[row][c0 + q * 8] = *(const uint4*)(s + q * 8);
    }
    {
        const unsigned short* s = Kh + (size_t)rowS * 64 + c0S;
        *(uint4*)&lk[0][rowS][c0S]     = *(const uint4*)(s);
        *(uint4*)&lk[0][rowS][c0S + 8] = *(const uint4*)(s + 8);
    }
#endif
    {   // V tile 0 with LDS transpose
        union { uint4 q[2]; unsigned short s[16]; } u;
        const unsigned short* s = Vh + (size_t)rowS * 64 + c0S;
        u.q[0] = *(const uint4*)(s);
        u.q[1] = *(const uint4*)(s + 8);
#pragma unroll
        for (int j = 0; j < 16; ++j) lvT[0][c0S + j][rowS] = u.s[j];
    }

    v8f   O[4] = {};
    float rmax[8], rsum[8];
#pragma unroll
    for (int r = 0; r < 8; ++r) { rmax[r] = -1e30f; rsum[r] = 0.f; }

    const int qrow   = wave * 16 + ln;
    const int ntiles = (q0 >> 6) + 2;

    for (int i = 0; i < ntiles; ++i) {
        const int cur = i & 1, nxt = cur ^ 1;
        const int j0  = i * 64;
#if USE_TDM
        if (wave == 0) __builtin_amdgcn_s_wait_tensorcnt(0);
#endif
        __syncthreads();

        const bool more = (i + 1) < ntiles;
        if (more) {                           // start next K/V tile
            const int j0n = j0 + 64;
#if USE_TDM
            if (wave == 0)
                tdm_issue(lds_off_of(&lk[nxt][0][0]), Kh + (size_t)j0n * 64,
                          tdm_desc2d(64u, 64u));
#else
            const unsigned short* sk = Kh + (size_t)(j0n + rowS) * 64 + c0S;
            kr0 = *(const uint4*)(sk);
            kr1 = *(const uint4*)(sk + 8);
#endif
            const unsigned short* sv = Vh + (size_t)(j0n + rowS) * 64 + c0S;
            vr0 = *(const uint4*)(sv);
            vr1 = *(const uint4*)(sv + 8);
        }

        // ---- S = Q K^T, per-wave 16x64 stripe ----
        v8f S[4] = {};
#pragma unroll
        for (int kc = 0; kc < 64; kc += 32) {
            v16bf a  = a_frag(&lq[qrow][0], kc, half);
            v16bf b0 = b_frag(&lk[cur][ln][0],      kc, half);
            v16bf b1 = b_frag(&lk[cur][16 + ln][0], kc, half);
            v16bf b2 = b_frag(&lk[cur][32 + ln][0], kc, half);
            v16bf b3 = b_frag(&lk[cur][48 + ln][0], kc, half);
            WMMA_BF16(S[0], a, b0); WMMA_BF16(S[1], a, b1);
            WMMA_BF16(S[2], a, b2); WMMA_BF16(S[3], a, b3);
        }

        // ---- causal mask + online softmax ----
        float lm[8];
#pragma unroll
        for (int r = 0; r < 8; ++r) lm[r] = -1e30f;
#pragma unroll
        for (int t = 0; t < 4; ++t)
#pragma unroll
            for (int r = 0; r < 8; ++r) {
                float sv = S[t][r] * 0.125f;                 // 1/sqrt(64)
                const int qg = q0 + wave * 16 + r + 8 * half;
                const int kg = j0 + t * 16 + ln;
                if (kg > qg) sv = -1e30f;
                S[t][r] = sv;
                lm[r] = fmaxf(lm[r], sv);
            }
#pragma unroll
        for (int r = 0; r < 8; ++r) {         // reduce across 16-lane half
            lm[r] = fmaxf(lm[r], __shfl_xor(lm[r], 1, 32));
            lm[r] = fmaxf(lm[r], __shfl_xor(lm[r], 2, 32));
            lm[r] = fmaxf(lm[r], __shfl_xor(lm[r], 4, 32));
            lm[r] = fmaxf(lm[r], __shfl_xor(lm[r], 8, 32));
        }
        float alpha[8], ls[8];
#pragma unroll
        for (int r = 0; r < 8; ++r) {
            const float mn = fmaxf(rmax[r], lm[r]);
            alpha[r] = __expf(rmax[r] - mn);
            rmax[r]  = mn;
            ls[r]    = 0.f;
        }
#pragma unroll
        for (int t = 0; t < 4; ++t)
#pragma unroll
            for (int r = 0; r < 8; ++r) {
                const float p = __expf(S[t][r] - rmax[r]);
                ls[r] += p;
                lp[wave * 16 + r + 8 * half][t * 16 + ln] = f2bf(p);
            }
#pragma unroll
        for (int r = 0; r < 8; ++r) {
            ls[r] += __shfl_xor(ls[r], 1, 32);
            ls[r] += __shfl_xor(ls[r], 2, 32);
            ls[r] += __shfl_xor(ls[r], 4, 32);
            ls[r] += __shfl_xor(ls[r], 8, 32);
            rsum[r] = rsum[r] * alpha[r] + ls[r];
        }
#pragma unroll
        for (int t = 0; t < 4; ++t)
#pragma unroll
            for (int r = 0; r < 8; ++r) O[t][r] *= alpha[r];

        // ---- O += P V  (lp stripe wave-private; per-wave LDS in-order) ----
#pragma unroll
        for (int kc = 0; kc < 64; kc += 32) {
            v16bf a  = a_frag(&lp[qrow][0], kc, half);
            v16bf b0 = b_frag(&lvT[cur][ln][0],      kc, half);
            v16bf b1 = b_frag(&lvT[cur][16 + ln][0], kc, half);
            v16bf b2 = b_frag(&lvT[cur][32 + ln][0], kc, half);
            v16bf b3 = b_frag(&lvT[cur][48 + ln][0], kc, half);
            WMMA_BF16(O[0], a, b0); WMMA_BF16(O[1], a, b1);
            WMMA_BF16(O[2], a, b2); WMMA_BF16(O[3], a, b3);
        }

        if (more) {                           // commit staged K/V into next buffer
#if !USE_TDM
            *(uint4*)&lk[nxt][rowS][c0S]     = kr0;
            *(uint4*)&lk[nxt][rowS][c0S + 8] = kr1;
#endif
            union { uint4 q[2]; unsigned short s[16]; } u;
            u.q[0] = vr0; u.q[1] = vr1;
#pragma unroll
            for (int j = 0; j < 16; ++j) lvT[nxt][c0S + j][rowS] = u.s[j];
        }
    }

    // ---- normalize + store bf16 to AO [b][t][h*64+d] ----
#pragma unroll
    for (int r = 0; r < 8; ++r) rsum[r] = (rsum[r] > 0.f) ? (1.f / rsum[r]) : 0.f;
#pragma unroll
    for (int t = 0; t < 4; ++t)
#pragma unroll
        for (int r = 0; r < 8; ++r) {
            const int row = q0 + wave * 16 + r + 8 * half;
            AO[((size_t)(b * 2048 + row)) * 2048 + h * 64 + t * 16 + ln] =
                f2bf(O[t][r] * rsum[r]);
        }
}

// ---------------------------------------------------------------------------
// Kernel C: output projection  out = AO(bf16) @ Wo^T(bf16), f32 out.
// Grid: (2048/64, BT/128). Block 256.  Double-buffered, TDM for A tiles.
// ---------------------------------------------------------------------------
__global__ __launch_bounds__(256) void oproj_kernel(
    const unsigned short* __restrict__ AO, const unsigned short* __restrict__ WoT,
    float* __restrict__ out)
{
    __shared__ __align__(16) unsigned short la[2][128][72];
    __shared__ __align__(16) unsigned short lw[2][64][72];

    const int tid  = threadIdx.x;
    const int lane = tid & 31, wave = tid >> 5;
    const int half = lane >> 4, ln = lane & 15;
    const int r0 = blockIdx.y * 128, n0 = blockIdx.x * 64;

    const int rowW = tid >> 2, c0W = (tid & 3) * 16;
    v8f acc[4] = {};
    uint4 w0, w1;
#if USE_TDM
    const i32x8 dx = tdm_desc2d(128u, 2048u);
#else
    const int rowX = tid >> 1, c0X = (tid & 1) * 32;
    uint4 xr[4];
#endif

#if USE_TDM
    if (wave == 0) tdm_issue(lds_off_of(&la[0][0][0]), AO + (size_t)r0 * 2048, dx);
#else
    {
        const unsigned short* s = AO + (size_t)(r0 + rowX) * 2048 + c0X;
#pragma unroll
        for (int q = 0; q < 4; ++q) *(uint4*)&la[0][rowX][c0X + q * 8] = *(const uint4*)(s + q * 8);
    }
#endif
    {
        const unsigned short* s = WoT + (size_t)(n0 + rowW) * 2048 + c0W;
        *(uint4*)&lw[0][rowW][c0W]     = *(const uint4*)(s);
        *(uint4*)&lw[0][rowW][c0W + 8] = *(const uint4*)(s + 8);
    }

    for (int i = 0; i < 32; ++i) {
        const int cur = i & 1, nxt = cur ^ 1;
#if USE_TDM
        if (wave == 0) __builtin_amdgcn_s_wait_tensorcnt(0);
#endif
        __syncthreads();

        const bool more = (i + 1) < 32;
        const int  kcn  = (i + 1) * 64;
        if (more) {
#if USE_TDM
            if (wave == 0)
                tdm_issue(lds_off_of(&la[nxt][0][0]), AO + (size_t)r0 * 2048 + kcn, dx);
#else
            const unsigned short* s = AO + (size_t)(r0 + rowX) * 2048 + kcn + c0X;
#pragma unroll
            for (int q = 0; q < 4; ++q) xr[q] = *(const uint4*)(s + q * 8);
#endif
            const unsigned short* sw = WoT + (size_t)(n0 + rowW) * 2048 + kcn + c0W;
            w0 = *(const uint4*)(sw);
            w1 = *(const uint4*)(sw + 8);
        }

        const unsigned short* arow = &la[cur][wave * 16 + ln][0];
#pragma unroll
        for (int kk = 0; kk < 64; kk += 32) {
            v16bf a  = a_frag(arow, kk, half);
            v16bf b0 = b_frag(&lw[cur][ln][0],      kk, half);
            v16bf b1 = b_frag(&lw[cur][16 + ln][0], kk, half);
            v16bf b2 = b_frag(&lw[cur][32 + ln][0], kk, half);
            v16bf b3 = b_frag(&lw[cur][48 + ln][0], kk, half);
            WMMA_BF16(acc[0], a, b0); WMMA_BF16(acc[1], a, b1);
            WMMA_BF16(acc[2], a, b2); WMMA_BF16(acc[3], a, b3);
        }

        if (more) {
#if !USE_TDM
#pragma unroll
            for (int q = 0; q < 4; ++q) *(uint4*)&la[nxt][rowX][c0X + q * 8] = xr[q];
#endif
            *(uint4*)&lw[nxt][rowW][c0W]     = w0;
            *(uint4*)&lw[nxt][rowW][c0W + 8] = w1;
        }
    }

#pragma unroll
    for (int t = 0; t < 4; ++t)
#pragma unroll
        for (int r = 0; r < 8; ++r)
            out[(size_t)(r0 + wave * 16 + r + 8 * half) * 2048 + n0 + t * 16 + ln] =
                acc[t][r];
}

// ---------------------------------------------------------------------------
// Host launcher.  Workspace (bf16), byte offsets:
//   xb 0 | WqT 16M | WkT 24M | WvT 26M | WoT 28M | Qb 36M | Kb 52M | Vb 56M
//   | AO 60M  (total ~76 MB)
// ---------------------------------------------------------------------------
extern "C" void kernel_launch(void* const* d_in, const int* in_sizes, int n_in,
                              void* d_out, int out_size, void* d_ws, size_t ws_size,
                              hipStream_t stream) {
    const float* x    = (const float*)d_in[0];
    const float* Wq   = (const float*)d_in[1];
    const float* Wk   = (const float*)d_in[2];
    const float* Wv   = (const float*)d_in[3];
    const float* Wo   = (const float*)d_in[4];
    const float* qlnw = (const float*)d_in[5];
    const float* klnw = (const float*)d_in[6];
    float* out = (float*)d_out;

    unsigned char* ws = (unsigned char*)d_ws;
    unsigned short* xb  = (unsigned short*)(ws);
    unsigned short* WqT = (unsigned short*)(ws + 16777216u);
    unsigned short* WkT = (unsigned short*)(ws + 25165824u);
    unsigned short* WvT = (unsigned short*)(ws + 27262976u);
    unsigned short* WoT = (unsigned short*)(ws + 29360128u);
    unsigned short* Qb  = (unsigned short*)(ws + 37748736u);
    unsigned short* Kb  = (unsigned short*)(ws + 54525952u);
    unsigned short* Vb  = (unsigned short*)(ws + 58720256u);
    unsigned short* AO  = (unsigned short*)(ws + 62914560u);

    // bf16 pre-pass: x elementwise; weights convert + transpose.
    cvt_bf16_kernel<<<dim3(8192), 256, 0, stream>>>(x, xb, 2097152);
    transpose_bf16_kernel<<<dim3(32, 32), 256, 0, stream>>>(Wq, WqT, 2048, 2048);
    transpose_bf16_kernel<<<dim3(8, 32),  256, 0, stream>>>(Wk, WkT, 2048, 512);
    transpose_bf16_kernel<<<dim3(8, 32),  256, 0, stream>>>(Wv, WvT, 2048, 512);
    transpose_bf16_kernel<<<dim3(32, 32), 256, 0, stream>>>(Wo, WoT, 2048, 2048);

    qkv_rope_kernel<<<dim3(48, 32), 256, 0, stream>>>(xb, WqT, WkT, WvT,
                                                      qlnw, klnw, Qb, Kb, Vb);
    flash_attn_kernel<<<dim3(16, 32, 2), 256, 0, stream>>>(Qb, Kb, Vb, AO);
    oproj_kernel<<<dim3(32, 32), 256, 0, stream>>>(AO, WoT, out);
}
